// PatchEmbed1w1a_78091095375874
// MI455X (gfx1250) — 8-bit matrix operand order repaired and checked by reading, compiled, not run on hardware
//
#include <hip/hip_runtime.h>
#include <hip/hip_bf16.h>


typedef __attribute__((ext_vector_type(8))) int v8i;

#define PATCHK   16
#define EMBED    768
#define KDIM     768
#define NP       14
#define PATCHES  196
#define BATCH    128
#define MTOT     (BATCH * PATCHES)
#define KSTEPS   (KDIM / 64)
#define HW       224

__device__ __forceinline__ signed char sgn8(float v) {
    return (signed char)((v > 0.0f) - (v < 0.0f));
}

__device__ __forceinline__ void async_ld16(unsigned lds_off, const void* gptr) {
    asm volatile("global_load_async_to_lds_b128 %0, %1, off"
                 :: "v"(lds_off), "v"(gptr)
                 : "memory");
}
__device__ __forceinline__ void wait_async0() {
    asm volatile("s_wait_asynccnt 0x0" ::: "memory");
}

__global__ void pe1w1a_prep_weights(const float* __restrict__ w,
                                    signed char* __restrict__ bsign,
                                    float* __restrict__ sf) {
    const int o   = blockIdx.x;
    const int tid = threadIdx.x;
    const float* wrow = w + (size_t)o * KDIM;
    signed char* brow = bsign + (size_t)o * KDIM;

    float s = 0.0f;
#pragma unroll
    for (int c = 0; c < 3; ++c) {
        float v = wrow[c * 256 + tid];
        s += fabsf(v);
        brow[c * 256 + tid] = sgn8(v);
    }

    __shared__ float red[256];
    red[tid] = s;
    __syncthreads();
#pragma unroll
    for (int off = 128; off > 0; off >>= 1) {
        if (tid < off) red[tid] += red[tid + off];
        __syncthreads();
    }
    if (tid == 0) sf[o] = red[0] * (1.0f / (float)KDIM);
}

__global__ void pe1w1a_im2col_sign(const float* __restrict__ x,
                                   const float* __restrict__ mb,
                                   signed char* __restrict__ A) {
    const int m   = blockIdx.x;
    const int tid = threadIdx.x;
    const int b  = m / PATCHES;
    const int p  = m - b * PATCHES;
    const int ph = p / NP;
    const int pw = p - ph * NP;
    const int kh = tid >> 4;
    const int kw = tid & 15;
    const int h  = ph * PATCHK - 1 + kh;
    const int wc = pw * PATCHK - 1 + kw;
    const bool in = (h >= 0) & (h < HW) & (wc >= 0) & (wc < HW);

    const size_t base = (size_t)b * 3 * HW * HW + (size_t)h * HW + wc;
    signed char* arow = A + (size_t)m * KDIM;
#pragma unroll
    for (int c = 0; c < 3; ++c) {
        signed char sv = 0;
        if (in) {
            float v = x[base + (size_t)c * HW * HW] + mb[c];
            sv = sgn8(v);
        }
        arow[c * 256 + tid] = sv;
    }
}

__global__ void __launch_bounds__(256)
pe1w1a_gemm_iu8(const signed char* __restrict__ Amat,
                const signed char* __restrict__ Bsign,
                const float* __restrict__ sf,
                float* __restrict__ out) {
    const int blk   = blockIdx.x;
    const int mTile = (blk / (EMBED / 128)) * 128;
    const int nTile = (blk % (EMBED / 128)) * 128;

    const int tid    = threadIdx.x;
    const int lane   = tid & 31;
    const int waveId = tid >> 5;
    const int waveN  = waveId & 1;
    const int waveM  = waveId >> 1;

    __shared__ __align__(16) signed char sA[2][128 * 64];
    __shared__ __align__(16) signed char sB[2][128 * 64];

    const unsigned ldsA = (unsigned)(uintptr_t)&sA[0][0];
    const unsigned ldsB = (unsigned)(uintptr_t)&sB[0][0];

    const int lrow   = tid >> 2;
    const int lcol16 = (tid & 3) << 4;

    v8i acc[2][4];
#pragma unroll
    for (int fm = 0; fm < 2; ++fm)
#pragma unroll
        for (int fn = 0; fn < 4; ++fn)
            acc[fm][fn] = (v8i){};

    const int laneRow = lane & 15;
    const int aKoff   = (lane >> 4) << 3;
    const int bChunk  = (lane >> 4) << 3;

    auto issue_tile = [&](int buf, int ki) {
        const int k0 = ki * 64;
#pragma unroll
        for (int phse = 0; phse < 2; ++phse) {
            const int r = phse * 64 + lrow;
            const unsigned loff = (unsigned)(buf * (128 * 64) + r * 64 + lcol16);
            async_ld16(ldsA + loff,
                       Amat + (size_t)(mTile + r) * KDIM + k0 + lcol16);
            async_ld16(ldsB + loff,
                       Bsign + (size_t)(nTile + r) * KDIM + k0 + lcol16);
        }
    };

    issue_tile(0, 0);

    for (int ki = 0; ki < KSTEPS; ++ki) {
        const int cur = ki & 1;

        wait_async0();
        __syncthreads();

        if (ki + 1 < KSTEPS) {
            issue_tile(cur ^ 1, ki + 1);
            __builtin_prefetch(Amat + (size_t)(mTile + lrow) * KDIM
                                   + (ki + 1) * 64 + lcol16, 0, 1);
        }

        v8i bfr[4];
#pragma unroll
        for (int fn = 0; fn < 4; ++fn) {
            const int col = waveN * 64 + fn * 16 + laneRow;
            const int* pb = (const int*)(&sB[cur][col * 64 + bChunk]);
            bfr[fn][0] = pb[0];  bfr[fn][1] = pb[1];
            bfr[fn][2] = pb[4];  bfr[fn][3] = pb[5];
            bfr[fn][4] = pb[8];  bfr[fn][5] = pb[9];
            bfr[fn][6] = pb[12]; bfr[fn][7] = pb[13];
        }

#pragma unroll
        for (int fm = 0; fm < 2; ++fm) {
            const int row = waveM * 32 + fm * 16 + laneRow;
            const int* pa = (const int*)(&sA[cur][row * 64 + aKoff]);
            v8i afr;
            afr[0] = pa[0];  afr[1] = pa[1];
            afr[2] = pa[4];  afr[3] = pa[5];
            afr[4] = pa[8];  afr[5] = pa[9];
            afr[6] = pa[12]; afr[7] = pa[13];
#pragma unroll
            for (int fn = 0; fn < 4; ++fn) {
                acc[fm][fn] = __builtin_amdgcn_wmma_i32_16x16x64_iu8(
                     true, afr,  true, bfr[fn],
                    acc[fm][fn],  false,  false);
            }
        }
    }

    float sc[4];
#pragma unroll
    for (int fn = 0; fn < 4; ++fn)
        sc[fn] = sf[nTile + waveN * 64 + fn * 16 + laneRow];

#pragma unroll
    for (int fm = 0; fm < 2; ++fm) {
        const int rowBase = mTile + waveM * 32 + fm * 16 + ((lane >> 4) << 3);
#pragma unroll
        for (int fn = 0; fn < 4; ++fn) {
            const int col = nTile + waveN * 64 + fn * 16 + laneRow;
#pragma unroll
            for (int r = 0; r < 8; ++r) {
                out[(size_t)(rowBase + r) * EMBED + col] =
                    (float)acc[fm][fn][r] * sc[fn];
            }
        }
    }
}

extern "C" void kernel_launch(void* const* d_in, const int* in_sizes, int n_in,
                              void* d_out, int out_size, void* d_ws, size_t ws_size,
                              hipStream_t stream) {
    const float* x  = (const float*)d_in[0];
    const float* w  = (const float*)d_in[1];
    const float* mb = (const float*)d_in[2];
    float* out = (float*)d_out;

    signed char* bsign = (signed char*)d_ws;
    float*       sf    = (float*)((char*)d_ws + (size_t)EMBED * KDIM);
    signed char* Amat  = (signed char*)((char*)d_ws + (size_t)EMBED * KDIM
                                        + (size_t)EMBED * sizeof(float));

    pe1w1a_prep_weights<<<EMBED, 256, 0, stream>>>(w, bsign, sf);
    pe1w1a_im2col_sign<<<MTOT, 256, 0, stream>>>(x, mb, Amat);

    const int grid = (MTOT / 128) * (EMBED / 128);
    pe1w1a_gemm_iu8<<<grid, 256, 0, stream>>>(Amat, bsign, sf, out);
}
